// NH1G3CSAttnRM_86431921864919
// MI455X (gfx1250) — matrix operand order repaired and checked by reading, not compiled, not run on hardware
//
#include <hip/hip_runtime.h>
#include <math.h>

// ---------------------------------------------------------------------------
// NH1G3CSAttnRM forward for MI455X (gfx1250, wave32).
// bf16 WMMA (v_wmma_f32_16x16x32_bf16) for all GEMMs; the dominant graph-conv
// GEMM (3600x7200x1152) uses CDNA5 async data movement: per-lane
// global_load_async_to_lds_b128 (ASYNCcnt) for the A tile and the Tensor Data
// Mover (tensor_load_to_lds, TENSORcnt) for the B tile, double-buffered so
// DMA overlaps the WMMA pipe. f32 VALU for softmax/layernorm/tiny MLPs.
// ---------------------------------------------------------------------------

typedef __bf16 bf16_t;
typedef __attribute__((ext_vector_type(16))) __bf16 bf16x16;
typedef __attribute__((ext_vector_type(8)))  float  f32x8;
typedef __attribute__((ext_vector_type(4)))  unsigned int u32x4_t;
typedef __attribute__((ext_vector_type(8)))  int i32x8_t;

#define FZ8 {0.f,0.f,0.f,0.f,0.f,0.f,0.f,0.f}

// ---------------- WMMA fragment helpers (wave32, 16x16x32 bf16) ------------
// A (MxK=16x32): lanes 0-15 -> M=lane, elem i -> K = i + half*8 + (i&8)
__device__ __forceinline__ bf16x16 ldA_bf16(const bf16_t* p, int ld, int krem) {
  int l = threadIdx.x & 31; int row = l & 15; int half = l >> 4;
  bf16x16 a;
#pragma unroll
  for (int i = 0; i < 16; ++i) {
    int k = i + half * 8 + (i & 8);
    a[i] = (k < krem) ? p[row * ld + k] : (bf16_t)0.0f;
  }
  return a;
}
__device__ __forceinline__ bf16x16 ldA_f32(const float* p, int ld, int krem) {
  int l = threadIdx.x & 31; int row = l & 15; int half = l >> 4;
  bf16x16 a;
#pragma unroll
  for (int i = 0; i < 16; ++i) {
    int k = i + half * 8 + (i & 8);
    a[i] = (k < krem) ? (bf16_t)p[row * ld + k] : (bf16_t)0.0f;
  }
  return a;
}
// B (KxN=32x16) row-major: elem i -> K = i + half*8 + (i&8), N = lane&15
__device__ __forceinline__ bf16x16 ldB_bf16(const bf16_t* p, int ld, int krem, int nrem) {
  int l = threadIdx.x & 31; int col = l & 15; int half = l >> 4;
  bf16x16 b;
#pragma unroll
  for (int i = 0; i < 16; ++i) {
    int k = i + half * 8 + (i & 8);
    b[i] = (k < krem && col < nrem) ? p[k * ld + col] : (bf16_t)0.0f;
  }
  return b;
}
// B^T: logical B[k][n] = src[n][k] (src row-major NxK) -- for Q @ K^T
__device__ __forceinline__ bf16x16 ldBT_bf16(const bf16_t* p, int ld, int krem) {
  int l = threadIdx.x & 31; int col = l & 15; int half = l >> 4;
  bf16x16 b;
#pragma unroll
  for (int i = 0; i < 16; ++i) {
    int k = i + half * 8 + (i & 8);
    b[i] = (k < krem) ? p[col * ld + k] : (bf16_t)0.0f;
  }
  return b;
}
__device__ __forceinline__ f32x8 wmma_bf16(bf16x16 a, bf16x16 b, f32x8 c) {
  return __builtin_amdgcn_wmma_f32_16x16x32_bf16(false, a, false, b, (short)0, c, false, false);
}

// ---------------- CDNA5 async data movement --------------------------------
// per-lane 16B memory -> LDS copy, tracked by ASYNCcnt
__device__ __forceinline__ void async_b128(unsigned int lds, const void* g) {
  asm volatile("global_load_async_to_lds_b128 %0, %1, off"
               :: "v"(lds), "v"((unsigned long long)(uintptr_t)g) : "memory");
}
__device__ __forceinline__ void wait_async0() {
  asm volatile("s_wait_asynccnt 0x0" ::: "memory");
}
// Tensor Data Mover: 2D bf16 tile (tile0 elems/row x tile1 rows, row stride
// stride0 elems) from global -> contiguous LDS.  D# per CDNA5 ISA section 8.
__device__ __forceinline__ void tdm_load_2d_bf16(unsigned int lds, const void* g,
                                                 int tile0, int tile1,
                                                 int stride0, int tdim0, int tdim1) {
  unsigned long long ga = (unsigned long long)(uintptr_t)g;
  u32x4_t g0;
  g0[0] = 1u;                                               // count=1 user D#
  g0[1] = lds;                                              // lds_addr
  g0[2] = (unsigned int)ga;                                 // global_addr[31:0]
  g0[3] = (unsigned int)((ga >> 32) & 0x01FFFFFFu) | 0x80000000u; // addr[56:32] | type=2
  i32x8_t g1;
  g1[0] = 0x00010000;                                       // data_size=1 (2 bytes)
  g1[1] = (int)(((unsigned)tdim0 & 0xFFFFu) << 16);         // tensor_dim0[15:0]
  g1[2] = (int)((((unsigned)tdim0) >> 16) | (((unsigned)tdim1 & 0xFFFFu) << 16));
  g1[3] = (int)((((unsigned)tdim1) >> 16) | (((unsigned)tile0) << 16)); // tile_dim0
  g1[4] = tile1 & 0xFFFF;                                   // tile_dim1 (tile_dim2=0)
  g1[5] = stride0;                                          // tensor_dim0_stride
  g1[6] = 0;
  g1[7] = 0;
  asm volatile("tensor_load_to_lds %0, %1" :: "s"(g0), "s"(g1) : "memory");
}

// ---------------- small scalar kernels -------------------------------------
__global__ void k_mlp2(const float* __restrict__ in, float* __restrict__ out,
                       const float* __restrict__ W1, const float* __restrict__ b1,
                       const float* __restrict__ W2, const float* __restrict__ b2,
                       int din, int dh, int dout) {
  extern __shared__ char smem[];
  float* hid = (float*)smem;
  int r = blockIdx.x, tid = threadIdx.x;
  const float* x = in + (size_t)r * din;
  for (int j = tid; j < dh; j += blockDim.x) {
    float a = b1[j];
    for (int i = 0; i < din; ++i) a += x[i] * W1[i * dh + j];
    hid[j] = fmaxf(a, 0.f);
  }
  __syncthreads();
  for (int j = tid; j < dout; j += blockDim.x) {
    float a = b2[j];
    for (int i = 0; i < dh; ++i) a += hid[i] * W2[i * dout + j];
    out[(size_t)r * dout + j] = fmaxf(a, 0.f);
  }
}

__global__ void k_lin(const float* __restrict__ in, float* __restrict__ out,
                      const float* __restrict__ W, const float* __restrict__ b,
                      int din, int dout) {
  int r = blockIdx.x, tid = threadIdx.x;
  const float* x = in + (size_t)r * din;
  for (int j = tid; j < dout; j += blockDim.x) {
    float a = b[j];
    for (int i = 0; i < din; ++i) a += x[i] * W[i * dout + j];
    out[(size_t)r * dout + j] = fmaxf(a, 0.f);
  }
}

__global__ void k_embMM(const float* __restrict__ mv, const float* __restrict__ msv,
                        const float* __restrict__ W, const float* __restrict__ b,
                        float* __restrict__ out) {
  int r = blockIdx.x, tid = threadIdx.x;   // r = b*150 + s
  int s = r % 150;
  if (tid < 18) {
    float a = b[tid];
    for (int i = 0; i < 72; ++i) a += mv[r * 72 + i] * W[i * 18 + tid];
    for (int i = 0; i < 72; ++i) a += msv[s * 72 + i] * W[(72 + i) * 18 + tid];
    out[r * 18 + tid] = fmaxf(a, 0.f);
  }
}

__global__ void k_f2bf(const float* __restrict__ src, bf16_t* __restrict__ dst, int n) {
  int i = blockIdx.x * 256 + threadIdx.x;
  if (i < n) dst[i] = (bf16_t)src[i];
}

// cc (b,c,t,d) f32 -> B matrix (7200 x 1152) bf16 with col = b*72+d
__global__ void k_pack_ccb(const float* __restrict__ cc, bf16_t* __restrict__ ccb) {
  size_t idx = (size_t)blockIdx.x * 256 + threadIdx.x;
  if (idx >= (size_t)7200 * 1152) return;
  int rowc = (int)(idx / 1152), col = (int)(idx % 1152);
  int b = col / 72, d = col % 72;
  ccb[idx] = (bf16_t)cc[((size_t)b * 7200 + rowc) * 72 + d];
}

// masked row softmax of GCW -> bf16 scores (3600 x 7200)
__global__ void k_scores(const float* __restrict__ GCW, const int* __restrict__ cmask,
                         bf16_t* __restrict__ S) {
  __shared__ float red[256];
  int i = blockIdx.x, tid = threadIdx.x;
  int si = i % 150;
  const float* row = GCW + (size_t)i * 7200;
  const int* cm = cmask + si * 300;
  float mx = -1e30f;
  for (int j = tid; j < 7200; j += 256) {
    float v = (cm[j % 300] == 1) ? -1e9f : row[j];
    mx = fmaxf(mx, v);
  }
  red[tid] = mx; __syncthreads();
  for (int s = 128; s > 0; s >>= 1) { if (tid < s) red[tid] = fmaxf(red[tid], red[tid + s]); __syncthreads(); }
  mx = red[0]; __syncthreads();
  float sm = 0.f;
  for (int j = tid; j < 7200; j += 256) {
    float v = (cm[j % 300] == 1) ? -1e9f : row[j];
    sm += __expf(v - mx);
  }
  red[tid] = sm; __syncthreads();
  for (int s = 128; s > 0; s >>= 1) { if (tid < s) red[tid] += red[tid + s]; __syncthreads(); }
  float inv = 1.f / red[0];
  for (int j = tid; j < 7200; j += 256) {
    float v = (cm[j % 300] == 1) ? -1e9f : row[j];
    S[(size_t)i * 7200 + j] = (bf16_t)(__expf(v - mx) * inv);
  }
}

// ---------------- graph-conv GEMM: (3600x7200)@(7200x1152) + GCB -----------
// Double-buffered: A tile via per-lane async-to-LDS b128, B tile via TDM.
__global__ void k_gemm_graph(const bf16_t* __restrict__ S, const bf16_t* __restrict__ B,
                             const float* __restrict__ GCB, float* __restrict__ out) {
  __shared__ __attribute__((aligned(16))) bf16_t As[2][48 * 32];
  __shared__ __attribute__((aligned(16))) bf16_t Bs[2][32 * 128];
  int tid = threadIdx.x, wave = tid >> 5, lane = tid & 31, l15 = lane & 15, half = lane >> 4;
  int m0 = blockIdx.x * 48, n0 = blockIdx.y * 128;
  f32x8 acc[3] = {FZ8, FZ8, FZ8};

  auto issue = [&](int buf, int k0) {
    if (tid < 192) {                 // A: 48 rows x 64B = 192 x 16B chunks
      int row = tid >> 2, c4 = tid & 3;
      async_b128((unsigned int)(uintptr_t)&As[buf][row * 32 + c4 * 8],
                 S + (size_t)(m0 + row) * 7200 + k0 + c4 * 8);
    }
    if (wave == 0) {                 // B: TDM 32 rows x 128 bf16, stride 1152
      tdm_load_2d_bf16((unsigned int)(uintptr_t)&Bs[buf][0],
                       B + (size_t)k0 * 1152 + n0,
                       128, 32, 1152, 1152, 7200);
    }
  };

  issue(0, 0);
  int cur = 0;
  for (int k0 = 0; k0 < 7200; k0 += 32) {
    wait_async0();
    __builtin_amdgcn_s_wait_tensorcnt(0);
    __syncthreads();                 // buffers for this iteration are ready
    if (k0 + 32 < 7200) issue(cur ^ 1, k0 + 32);   // overlap next DMA w/ WMMA
    bf16x16 bf = ldB_bf16(Bs[cur] + wave * 16, 128, 32, 128);
#pragma unroll
    for (int mt = 0; mt < 3; ++mt)
      acc[mt] = wmma_bf16(ldA_bf16(As[cur] + mt * 16 * 32, 32, 32), bf, acc[mt]);
    cur ^= 1;
  }
  int coln = n0 + wave * 16 + l15;
  int bb = coln / 72, d = coln % 72;
#pragma unroll
  for (int mt = 0; mt < 3; ++mt) {
#pragma unroll
    for (int r = 0; r < 8; ++r) {
      int row = m0 + mt * 16 + half * 8 + r;
      out[((size_t)bb * 3600 + row) * 72 + d] = acc[mt][r] + GCB[row * 72 + d];
    }
  }
}

// ---------------- attention blocks -----------------------------------------
struct AttnW {
  const bf16_t *Wq, *Wk, *Wv, *Wo, *F1, *F2;
  const float *bq, *bk, *bv, *bo, *a1, *g1, *a2, *g2, *fb1, *fb2;
};

__device__ __forceinline__ float emb_CT(int b, int cidx, int t, int d,
                                        const float* embCe, const float* embTe, const float* embPe) {
  if (d < 36) return embCe[cidx * 36 + d];
  if (d < 54) return embTe[t * 18 + d - 36];
  return embPe[b * 18 + d - 54];
}
__device__ __forceinline__ float emb_MMST(int b, int s, int t, int d,
                                          const float* embMM, const float* embS,
                                          const float* embTe, const float* embPe) {
  if (d < 18) return embMM[(b * 150 + s) * 18 + d];
  if (d < 36) return embS[s * 18 + d - 18];
  if (d < 54) return embTe[t * 18 + d - 36];
  return embPe[b * 18 + d - 54];
}

// Temporal self-attention block: one workgroup per (b, n), seq length T=24.
// LDS (dynamic, 34816 B): Xf f32[32][80] | Hb bf16[32][80] | QKV bf16 3x[32][80] | Att f32[32][32]
__global__ void k_attn_temporal(const float* __restrict__ xin, float* __restrict__ xout,
                                int N, int mode,
                                const float* embA, const float* embB,
                                const float* embTe, const float* embPe, AttnW w) {
  extern __shared__ char smem[];
  float*  Xf  = (float*)smem;                 // 10240
  bf16_t* Hb  = (bf16_t*)(smem + 10240);      // 5120
  bf16_t* Qb  = (bf16_t*)(smem + 15360);      // 3x5120
  bf16_t* Kb  = Qb + 32 * 80;
  bf16_t* Vb  = Kb + 32 * 80;
  float*  Att = (float*)(smem + 30720);       // 4096
  bf16_t* Y1  = Qb;                           // [32][256] aliases QKV+Att (FFN phase)

  int tid = threadIdx.x, wave = tid >> 5, lane = tid & 31, l15 = lane & 15, half = lane >> 4;
  int bi = blockIdx.x / N, n = blockIdx.x % N;
  const int T = 24;
  size_t base = ((size_t)(bi * N + n)) * T * 72;

  for (int i = tid; i < 32 * 80; i += blockDim.x) {
    int r = i / 80, c = i % 80;
    float v = (r < T && c < 72) ? xin[base + r * 72 + c] : 0.f;
    Xf[i] = v;
    if (r < T && c < 72) {
      float e = (mode == 0) ? emb_CT(bi, n, r, c, embA, embTe, embPe)
                            : emb_MMST(bi, n, r, c, embA, embB, embTe, embPe);
      Hb[i] = (bf16_t)(v + e);
    } else Hb[i] = (bf16_t)0.0f;
  }
  __syncthreads();

  // Q,K,V = H @ W + b
  const bf16_t* Wm[3] = {w.Wq, w.Wk, w.Wv};
  const float*  Bm[3] = {w.bq, w.bk, w.bv};
  bf16_t* Dst[3] = {Qb, Kb, Vb};
  for (int p = 0; p < 3; ++p) {
    for (int job = wave; job < 10; job += 8) {
      int mt = job / 5, nt = job % 5;
      f32x8 acc = FZ8;
      for (int k0 = 0; k0 < 72; k0 += 32) {
        bf16x16 a = ldA_bf16(Hb + mt * 16 * 80 + k0, 80, 72 - k0);
        bf16x16 b = ldB_bf16(Wm[p] + k0 * 72 + nt * 16, 72, 72 - k0, 72 - nt * 16);
        acc = wmma_bf16(a, b, acc);
      }
      int col = nt * 16 + l15;
      float bias = (col < 72) ? Bm[p][col] : 0.f;
#pragma unroll
      for (int r = 0; r < 8; ++r) {
        int row = mt * 16 + half * 8 + r;
        Dst[p][row * 80 + col] = (col < 72) ? (bf16_t)(acc[r] + bias) : (bf16_t)0.0f;
      }
    }
  }
  __syncthreads();

  // scores = Q K^T / sqrt(72)
  const float scale = 0.11785113f;
  for (int job = wave; job < 4; job += 8) {
    int mt = job >> 1, nt = job & 1;
    f32x8 acc = FZ8;
    for (int k0 = 0; k0 < 72; k0 += 32) {
      bf16x16 a = ldA_bf16(Qb + mt * 16 * 80 + k0, 80, 72 - k0);
      bf16x16 b = ldBT_bf16(Kb + nt * 16 * 80 + k0, 80, 72 - k0);
      acc = wmma_bf16(a, b, acc);
    }
#pragma unroll
    for (int r = 0; r < 8; ++r)
      Att[(mt * 16 + half * 8 + r) * 32 + nt * 16 + l15] = acc[r] * scale;
  }
  __syncthreads();
  if (tid < 32) {
    float* row = Att + tid * 32;
    if (tid < T) {
      float mx = -1e30f;
      for (int j = 0; j < T; ++j) mx = fmaxf(mx, row[j]);
      float sm = 0.f;
      for (int j = 0; j < T; ++j) { float e = __expf(row[j] - mx); row[j] = e; sm += e; }
      float inv = 1.f / sm;
      for (int j = 0; j < T; ++j) row[j] *= inv;
      for (int j = T; j < 32; ++j) row[j] = 0.f;
    } else for (int j = 0; j < 32; ++j) row[j] = 0.f;
  }
  __syncthreads();

  // O1 = Att @ V -> Hb (reuse)
  for (int job = wave; job < 10; job += 8) {
    int mt = job / 5, nt = job % 5;
    f32x8 acc = FZ8;
    bf16x16 a = ldA_f32(Att + mt * 16 * 32, 32, 32);
    bf16x16 b = ldB_bf16(Vb + nt * 16, 80, 32, 80);
    acc = wmma_bf16(a, b, acc);
#pragma unroll
    for (int r = 0; r < 8; ++r)
      Hb[(mt * 16 + half * 8 + r) * 80 + nt * 16 + l15] = (bf16_t)acc[r];
  }
  __syncthreads();

  // y = x + O1 @ Wo + bo
  for (int job = wave; job < 10; job += 8) {
    int mt = job / 5, nt = job % 5;
    f32x8 acc = FZ8;
    for (int k0 = 0; k0 < 72; k0 += 32) {
      bf16x16 a = ldA_bf16(Hb + mt * 16 * 80 + k0, 80, 72 - k0);
      bf16x16 b = ldB_bf16(w.Wo + k0 * 72 + nt * 16, 72, 72 - k0, 72 - nt * 16);
      acc = wmma_bf16(a, b, acc);
    }
    int col = nt * 16 + l15;
    if (col < 72) {
      float bias = w.bo[col];
#pragma unroll
      for (int r = 0; r < 8; ++r)
        Xf[(mt * 16 + half * 8 + r) * 80 + col] += acc[r] + bias;
    }
  }
  __syncthreads();

  // norm1
  if (tid < T) {
    float* row = Xf + tid * 80;
    float mu = 0.f; for (int j = 0; j < 72; ++j) mu += row[j]; mu *= (1.f / 72.f);
    float var = 0.f; for (int j = 0; j < 72; ++j) { float d = row[j] - mu; var += d * d; }
    float inv = 1.f / (sqrtf(var * (1.f / 72.f)) + 1e-6f);
    for (int j = 0; j < 72; ++j) {
      float v = w.a1[j] * (row[j] - mu) * inv + w.g1[j];
      row[j] = v;
      Hb[tid * 80 + j] = (bf16_t)v;
    }
  }
  __syncthreads();

  // FFN1: Y1 = relu(y @ F1 + fb1)   [32][256]
  for (int job = wave; job < 32; job += 8) {
    int mt = job / 16, nt = job % 16;
    f32x8 acc = FZ8;
    for (int k0 = 0; k0 < 72; k0 += 32) {
      bf16x16 a = ldA_bf16(Hb + mt * 16 * 80 + k0, 80, 72 - k0);
      bf16x16 b = ldB_bf16(w.F1 + k0 * 256 + nt * 16, 256, 72 - k0, 256);
      acc = wmma_bf16(a, b, acc);
    }
    int col = nt * 16 + l15;
    float bias = w.fb1[col];
#pragma unroll
    for (int r = 0; r < 8; ++r)
      Y1[(mt * 16 + half * 8 + r) * 256 + col] = (bf16_t)fmaxf(acc[r] + bias, 0.f);
  }
  __syncthreads();

  // FFN2 + residual
  for (int job = wave; job < 10; job += 8) {
    int mt = job / 5, nt = job % 5;
    f32x8 acc = FZ8;
    for (int k0 = 0; k0 < 256; k0 += 32) {
      bf16x16 a = ldA_bf16(Y1 + mt * 16 * 256 + k0, 256, 32);
      bf16x16 b = ldB_bf16(w.F2 + k0 * 72 + nt * 16, 72, 32, 72 - nt * 16);
      acc = wmma_bf16(a, b, acc);
    }
    int col = nt * 16 + l15;
    if (col < 72) {
      float bias = w.fb2[col];
#pragma unroll
      for (int r = 0; r < 8; ++r)
        Xf[(mt * 16 + half * 8 + r) * 80 + col] += fmaxf(acc[r] + bias, 0.f);
    }
  }
  __syncthreads();

  // norm2 -> out
  if (tid < T) {
    float* row = Xf + tid * 80;
    float mu = 0.f; for (int j = 0; j < 72; ++j) mu += row[j]; mu *= (1.f / 72.f);
    float var = 0.f; for (int j = 0; j < 72; ++j) { float d = row[j] - mu; var += d * d; }
    float inv = 1.f / (sqrtf(var * (1.f / 72.f)) + 1e-6f);
    for (int j = 0; j < 72; ++j)
      xout[base + tid * 72 + j] = w.a2[j] * (row[j] - mu) * inv + w.g2[j];
  }
}

// Spatial self-attention block: one workgroup per (b, t), seq length N=150.
// LDS (174080 B): Xf f32[160][80] | Hb bf16[160][80] | QKV bf16 3x[160][80] | Att f32[32][160]
__global__ void k_attn_spatial(const float* __restrict__ xin, float* __restrict__ xout,
                               const float* embMM, const float* embS,
                               const float* embTe, const float* embPe, AttnW w) {
  extern __shared__ char smem[];
  float*  Xf  = (float*)smem;                  // 51200
  bf16_t* Hb  = (bf16_t*)(smem + 51200);       // 25600
  bf16_t* Qb  = (bf16_t*)(smem + 76800);       // 3x25600
  bf16_t* Kb  = Qb + 160 * 80;
  bf16_t* Vb  = Kb + 160 * 80;
  float*  Att = (float*)(smem + 153600);       // 20480 (32x160)
  bf16_t* Y1  = Qb;                            // [160][256] aliases QKV(+Att)

  int tid = threadIdx.x, wave = tid >> 5, lane = tid & 31, l15 = lane & 15, half = lane >> 4;
  int bi = blockIdx.x / 24, t = blockIdx.x % 24;
  const int NS = 150;

  for (int i = tid; i < 160 * 80; i += blockDim.x) {
    int r = i / 80, c = i % 80;
    float v = (r < NS && c < 72) ? xin[(((size_t)(bi * 150 + r)) * 24 + t) * 72 + c] : 0.f;
    Xf[i] = v;
    Hb[i] = (r < NS && c < 72)
              ? (bf16_t)(v + emb_MMST(bi, r, t, c, embMM, embS, embTe, embPe))
              : (bf16_t)0.0f;
  }
  __syncthreads();

  const bf16_t* Wm[3] = {w.Wq, w.Wk, w.Wv};
  const float*  Bm[3] = {w.bq, w.bk, w.bv};
  bf16_t* Dst[3] = {Qb, Kb, Vb};
  for (int p = 0; p < 3; ++p) {
    for (int job = wave; job < 50; job += 8) {
      int mt = job / 5, nt = job % 5;
      f32x8 acc = FZ8;
      for (int k0 = 0; k0 < 72; k0 += 32) {
        bf16x16 a = ldA_bf16(Hb + mt * 16 * 80 + k0, 80, 72 - k0);
        bf16x16 b = ldB_bf16(Wm[p] + k0 * 72 + nt * 16, 72, 72 - k0, 72 - nt * 16);
        acc = wmma_bf16(a, b, acc);
      }
      int col = nt * 16 + l15;
      float bias = (col < 72) ? Bm[p][col] : 0.f;
#pragma unroll
      for (int r = 0; r < 8; ++r) {
        int row = mt * 16 + half * 8 + r;
        Dst[p][row * 80 + col] = (col < 72) ? (bf16_t)(acc[r] + bias) : (bf16_t)0.0f;
      }
    }
  }
  __syncthreads();

  const float scale = 0.11785113f;
  for (int mb = 0; mb < 5; ++mb) {          // 32-row attention blocks
    for (int job = wave; job < 20; job += 8) {
      int mt = job / 10, nt = job % 10;
      f32x8 acc = FZ8;
      for (int k0 = 0; k0 < 72; k0 += 32) {
        bf16x16 a = ldA_bf16(Qb + (mb * 32 + mt * 16) * 80 + k0, 80, 72 - k0);
        bf16x16 b = ldBT_bf16(Kb + nt * 16 * 80 + k0, 80, 72 - k0);
        acc = wmma_bf16(a, b, acc);
      }
#pragma unroll
      for (int r = 0; r < 8; ++r)
        Att[(mt * 16 + half * 8 + r) * 160 + nt * 16 + l15] = acc[r] * scale;
    }
    __syncthreads();
    if (tid < 32) {
      int gr = mb * 32 + tid;
      float* row = Att + tid * 160;
      if (gr < NS) {
        float mx = -1e30f;
        for (int j = 0; j < NS; ++j) mx = fmaxf(mx, row[j]);
        float sm = 0.f;
        for (int j = 0; j < NS; ++j) { float e = __expf(row[j] - mx); row[j] = e; sm += e; }
        float inv = 1.f / sm;
        for (int j = 0; j < NS; ++j) row[j] *= inv;
        for (int j = NS; j < 160; ++j) row[j] = 0.f;
      } else for (int j = 0; j < 160; ++j) row[j] = 0.f;
    }
    __syncthreads();
    for (int job = wave; job < 10; job += 8) {   // O block -> Hb rows
      int mt = job / 5, nt = job % 5;
      f32x8 acc = FZ8;
      for (int k0 = 0; k0 < 160; k0 += 32) {
        bf16x16 a = ldA_f32(Att + mt * 16 * 160 + k0, 160, 32);
        bf16x16 b = ldB_bf16(Vb + k0 * 80 + nt * 16, 80, 32, 80);
        acc = wmma_bf16(a, b, acc);
      }
#pragma unroll
      for (int r = 0; r < 8; ++r)
        Hb[(mb * 32 + mt * 16 + half * 8 + r) * 80 + nt * 16 + l15] = (bf16_t)acc[r];
    }
    __syncthreads();
  }

  for (int job = wave; job < 50; job += 8) {   // y = x + O @ Wo + bo
    int mt = job / 5, nt = job % 5;
    f32x8 acc = FZ8;
    for (int k0 = 0; k0 < 72; k0 += 32) {
      bf16x16 a = ldA_bf16(Hb + mt * 16 * 80 + k0, 80, 72 - k0);
      bf16x16 b = ldB_bf16(w.Wo + k0 * 72 + nt * 16, 72, 72 - k0, 72 - nt * 16);
      acc = wmma_bf16(a, b, acc);
    }
    int col = nt * 16 + l15;
    if (col < 72) {
      float bias = w.bo[col];
#pragma unroll
      for (int r = 0; r < 8; ++r)
        Xf[(mt * 16 + half * 8 + r) * 80 + col] += acc[r] + bias;
    }
  }
  __syncthreads();
  for (int rr = tid; rr < NS; rr += blockDim.x) {   // norm1
    float* row = Xf + rr * 80;
    float mu = 0.f; for (int j = 0; j < 72; ++j) mu += row[j]; mu *= (1.f / 72.f);
    float var = 0.f; for (int j = 0; j < 72; ++j) { float d = row[j] - mu; var += d * d; }
    float inv = 1.f / (sqrtf(var * (1.f / 72.f)) + 1e-6f);
    for (int j = 0; j < 72; ++j) {
      float v = w.a1[j] * (row[j] - mu) * inv + w.g1[j];
      row[j] = v;
      Hb[rr * 80 + j] = (bf16_t)v;
    }
  }
  __syncthreads();
  for (int job = wave; job < 160; job += 8) {  // FFN1 -> Y1 [160][256]
    int mt = job / 16, nt = job % 16;
    f32x8 acc = FZ8;
    for (int k0 = 0; k0 < 72; k0 += 32) {
      bf16x16 a = ldA_bf16(Hb + mt * 16 * 80 + k0, 80, 72 - k0);
      bf16x16 b = ldB_bf16(w.F1 + k0 * 256 + nt * 16, 256, 72 - k0, 256);
      acc = wmma_bf16(a, b, acc);
    }
    int col = nt * 16 + l15;
    float bias = w.fb1[col];
#pragma unroll
    for (int r = 0; r < 8; ++r)
      Y1[(mt * 16 + half * 8 + r) * 256 + col] = (bf16_t)fmaxf(acc[r] + bias, 0.f);
  }
  __syncthreads();
  for (int job = wave; job < 50; job += 8) {   // FFN2 + residual
    int mt = job / 5, nt = job % 5;
    f32x8 acc = FZ8;
    for (int k0 = 0; k0 < 256; k0 += 32) {
      bf16x16 a = ldA_bf16(Y1 + mt * 16 * 256 + k0, 256, 32);
      bf16x16 b = ldB_bf16(w.F2 + k0 * 72 + nt * 16, 72, 32, 72 - nt * 16);
      acc = wmma_bf16(a, b, acc);
    }
    int col = nt * 16 + l15;
    if (col < 72) {
      float bias = w.fb2[col];
#pragma unroll
      for (int r = 0; r < 8; ++r)
        Xf[(mt * 16 + half * 8 + r) * 80 + col] += fmaxf(acc[r] + bias, 0.f);
    }
  }
  __syncthreads();
  for (int rr = tid; rr < NS; rr += blockDim.x) {   // norm2 -> out
    float* row = Xf + rr * 80;
    float mu = 0.f; for (int j = 0; j < 72; ++j) mu += row[j]; mu *= (1.f / 72.f);
    float var = 0.f; for (int j = 0; j < 72; ++j) { float d = row[j] - mu; var += d * d; }
    float inv = 1.f / (sqrtf(var * (1.f / 72.f)) + 1e-6f);
    for (int j = 0; j < 72; ++j)
      xout[(((size_t)(bi * 150 + rr)) * 24 + t) * 72 + j] = w.a2[j] * (row[j] - mu) * inv + w.g2[j];
  }
}

// ---------------- final MLP: concat(x, embMMST)(144) -> 256 -> 1 -----------
__global__ void k_final(const float* __restrict__ x, const float* __restrict__ embMM,
                        const float* __restrict__ embS, const float* __restrict__ embTe,
                        const float* __restrict__ embPe,
                        const float* __restrict__ W1, const float* __restrict__ b1,
                        const float* __restrict__ W2, const float* __restrict__ b2,
                        float* __restrict__ out) {
  __shared__ float hid[256];
  __shared__ float feat[144];
  int r = blockIdx.x, tid = threadIdx.x;
  int b = r / 3600, rem = r % 3600, s = rem / 24, t = rem % 24;
  if (tid < 144) {
    float v;
    if (tid < 72)       v = x[(size_t)r * 72 + tid];
    else if (tid < 90)  v = embMM[(b * 150 + s) * 18 + tid - 72];
    else if (tid < 108) v = embS[s * 18 + tid - 90];
    else if (tid < 126) v = embTe[t * 18 + tid - 108];
    else                v = embPe[b * 18 + tid - 126];
    feat[tid] = v;
  }
  __syncthreads();
  float a = b1[tid];
  for (int i = 0; i < 144; ++i) a += feat[i] * W1[i * 256 + tid];
  hid[tid] = fmaxf(a, 0.f) * W2[tid];
  __syncthreads();
  for (int s2 = 128; s2 > 0; s2 >>= 1) { if (tid < s2) hid[tid] += hid[tid + s2]; __syncthreads(); }
  if (tid == 0) out[r] = fmaxf(hid[0] + b2[0], 0.f);
}

// ---------------------------------------------------------------------------
static inline AttnW mkW(void* const* d_in, int ib, bf16_t* wb) {
  AttnW w;
  w.Wq = wb; w.Wk = wb + 5184; w.Wv = wb + 10368; w.Wo = wb + 15552;
  w.F1 = wb + 20736; w.F2 = wb + 39168;
  w.bq  = (const float*)d_in[ib + 1];  w.bk = (const float*)d_in[ib + 3];
  w.bv  = (const float*)d_in[ib + 5];  w.bo = (const float*)d_in[ib + 7];
  w.a1  = (const float*)d_in[ib + 8];  w.g1 = (const float*)d_in[ib + 9];
  w.a2  = (const float*)d_in[ib + 10]; w.g2 = (const float*)d_in[ib + 11];
  w.fb1 = (const float*)d_in[ib + 13]; w.fb2 = (const float*)d_in[ib + 15];
  return w;
}

extern "C" void kernel_launch(void* const* d_in, const int* in_sizes, int n_in,
                              void* d_out, int out_size, void* d_ws, size_t ws_size,
                              hipStream_t stream) {
  (void)in_sizes; (void)n_in; (void)out_size; (void)ws_size;
  auto F = [&](int i) { return (const float*)d_in[i]; };
  const float* m_in  = F(0);
  const float* c_in  = F(1);
  const float* p_in  = F(2);
  const int*   cmask = (const int*)d_in[3];

  char* ws = (char*)d_ws;
  size_t off = 0;
  auto alloc = [&](size_t bytes) { size_t o = off; off += (bytes + 255) & ~(size_t)255; return o; };
  size_t o_mv    = alloc(2400 * 72 * 4);
  size_t o_msv   = alloc(150 * 72 * 4);
  size_t o_embMM = alloc(2400 * 18 * 4);
  size_t o_embS  = alloc(150 * 18 * 4);
  size_t o_embT  = alloc(24 * 18 * 4);
  size_t o_embP  = alloc(16 * 18 * 4);
  size_t o_embC  = alloc(300 * 36 * 4);
  size_t o_wbf   = alloc(4 * 57600 * 2);
  size_t o_cc    = alloc((size_t)16 * 300 * 24 * 72 * 4);   // 33.2 MB (reused as xa|xb later)
  size_t o_ccb   = alloc((size_t)7200 * 1152 * 2);          // 16.6 MB
  size_t o_sc    = alloc((size_t)3600 * 7200 * 2);          // 51.8 MB
  size_t o_s     = alloc((size_t)16 * 3600 * 72 * 4);       // 16.6 MB

  float*  mv     = (float*)(ws + o_mv);
  float*  msv    = (float*)(ws + o_msv);
  float*  embMM  = (float*)(ws + o_embMM);
  float*  embS   = (float*)(ws + o_embS);
  float*  embTe  = (float*)(ws + o_embT);
  float*  embPe  = (float*)(ws + o_embP);
  float*  embCe  = (float*)(ws + o_embC);
  bf16_t* wbf    = (bf16_t*)(ws + o_wbf);
  float*  ccf    = (float*)(ws + o_cc);
  bf16_t* ccb    = (bf16_t*)(ws + o_ccb);
  bf16_t* scores = (bf16_t*)(ws + o_sc);
  float*  sbuf   = (float*)(ws + o_s);
  float*  xa     = (float*)(ws + o_cc);                     // reuse cc region
  float*  xb     = (float*)(ws + o_cc + (size_t)16 * 3600 * 72 * 4);

  // small embedding tables
  k_mlp2<<<2400, 128, 72 * 4, stream>>>(m_in, mv, F(4), F(5), F(6), F(7), 10, 72, 72);
  k_mlp2<<<150, 128, 72 * 4, stream>>>(F(101), msv, F(8), F(9), F(10), F(11), 9, 72, 72);
  k_mlp2<<<150, 128, 144 * 4, stream>>>(F(100), embS, F(18), F(19), F(20), F(21), 64, 144, 18);
  k_lin<<<24, 32, 0, stream>>>(F(35), embTe, F(26), F(27), 72, 18);
  k_lin<<<16, 32, 0, stream>>>(p_in, embPe, F(24), F(25), 8, 18);
  k_lin<<<300, 64, 0, stream>>>(F(34), embCe, F(22), F(23), 72, 36);
  k_embMM<<<2400, 32, 0, stream>>>(mv, msv, F(16), F(17), embMM);

  // cc = mlp2(c)
  k_mlp2<<<115200, 128, 72 * 4, stream>>>(c_in, ccf, F(12), F(13), F(14), F(15), 2, 72, 72);

  // bf16 weight copies for the 4 attention blocks
  const int woff[6] = {0, 5184, 10368, 15552, 20736, 39168};
  const int widx[6] = {0, 2, 4, 6, 12, 14};
  const int wsz[6]  = {5184, 5184, 5184, 5184, 18432, 18432};
  for (int k = 0; k < 4; ++k) {
    int ib = 36 + 16 * k;
    bf16_t* wb = wbf + (size_t)k * 57600;
    for (int j = 0; j < 6; ++j)
      k_f2bf<<<(wsz[j] + 255) / 256, 256, 0, stream>>>(F(ib + widx[j]), wb + woff[j], wsz[j]);
  }
  AttnW wTC = mkW(d_in, 36, wbf + 0 * 57600);
  AttnW wTS = mkW(d_in, 52, wbf + 1 * 57600);
  AttnW wT1 = mkW(d_in, 68, wbf + 2 * 57600);
  AttnW wS1 = mkW(d_in, 84, wbf + 3 * 57600);

  const size_t SMEM_T = 34816, SMEM_S = 174080;

  // cc = attnTC(cc, embCT)  (temporal, N=300, in-place per (b,n) slice)
  k_attn_temporal<<<16 * 300, 256, SMEM_T, stream>>>(ccf, ccf, 300, 0, embCe, nullptr, embTe, embPe, wTC);

  // graph conv: s = softmax(masked GCW) @ cc + GCB
  k_pack_ccb<<<(int)(((size_t)7200 * 1152 + 255) / 256), 256, 0, stream>>>(ccf, ccb);
  k_scores<<<3600, 256, 0, stream>>>(F(32), cmask, scores);
  dim3 gg(75, 9);
  k_gemm_graph<<<gg, 256, 0, stream>>>(scores, ccb, F(33), sbuf);

  // s = attnTS(s); x = attnT1(x); x = attnS1(x, spatial)
  k_attn_temporal<<<16 * 150, 256, SMEM_T, stream>>>(sbuf, xa, 150, 1, embMM, embS, embTe, embPe, wTS);
  k_attn_temporal<<<16 * 150, 256, SMEM_T, stream>>>(xa, xb, 150, 1, embMM, embS, embTe, embPe, wT1);
  k_attn_spatial<<<16 * 24, 256, SMEM_S, stream>>>(xb, xa, embMM, embS, embTe, embPe, wS1);

  // final head
  k_final<<<57600, 256, 0, stream>>>(xa, embMM, embS, embTe, embPe,
                                     F(28), F(29), F(30), F(31), (float*)d_out);
}
